// RNNcell_20426864459845
// MI455X (gfx1250) — matrix operand order repaired and checked by reading, not compiled, not run on hardware
//
#include <hip/hip_runtime.h>

typedef __bf16 bf16_t;
typedef bf16_t v8bf  __attribute__((ext_vector_type(8)));
typedef bf16_t v16bf __attribute__((ext_vector_type(16)));
typedef float  v8f   __attribute__((ext_vector_type(8)));

#define IN_DIM  1024
#define HID_DIM 2048
#define OUT_DIM 1024
#define SEQ     256
#define BATCH   256

#define WMMA_BF16(A_, B_, C_) \
    __builtin_amdgcn_wmma_f32_16x16x32_bf16(false, (A_), false, (B_), (short)0, (C_), false, false)

__device__ __forceinline__ bf16_t f2bf(float f) {
    unsigned u = __builtin_bit_cast(unsigned, f);
    u += 0x7FFFu + ((u >> 16) & 1u);            // round-to-nearest-even
    unsigned short h = (unsigned short)(u >> 16);
    return __builtin_bit_cast(bf16_t, h);
}

__device__ __forceinline__ v16bf joinbf(v8bf lo, v8bf hi) {
    v16bf r;
#pragma unroll
    for (int i = 0; i < 8; ++i) { r[i] = lo[i]; r[8 + i] = hi[i]; }
    return r;
}

// B fragment: 16 contiguous bf16 (K-contiguous, k = 16*(lane>>4) + s, same order as A)
__device__ __forceinline__ v16bf loadFragB(const bf16_t* p) {
    return joinbf(*(const v8bf*)p, *(const v8bf*)(p + 8));
}
// A fragment: 16 contiguous K, [lo,lo+16), lo = (lane>>4)*16 (matches B)
__device__ __forceinline__ v16bf loadFragA(const bf16_t* p) {
    return joinbf(*(const v8bf*)p, *(const v8bf*)(p + 8));
}

// ---------------------------------------------------------------------------
// One-time conversion: U, V, OW -> bf16 copies; H_prev -> transposed bf16 HT1
// ---------------------------------------------------------------------------
__global__ __launch_bounds__(256)
void rnn_cvt_kernel(const float* __restrict__ U, const float* __restrict__ V,
                    const float* __restrict__ OW, const float* __restrict__ Hprev,
                    bf16_t* __restrict__ Ubf, bf16_t* __restrict__ Vbf,
                    bf16_t* __restrict__ OWbf, bf16_t* __restrict__ HT1)
{
    const int UN = HID_DIM * IN_DIM;     // 2M
    const int VN = HID_DIM * HID_DIM;    // 4M
    const int ON = OUT_DIM * HID_DIM;    // 2M
    const int HN = HID_DIM * BATCH;      // 512K
    const int total = UN + VN + ON + HN;
    for (int idx = blockIdx.x * 256 + threadIdx.x; idx < total; idx += gridDim.x * 256) {
        if (idx < UN) {
            Ubf[idx] = f2bf(U[idx]);
        } else if (idx < UN + VN) {
            int i = idx - UN; Vbf[i] = f2bf(V[i]);
        } else if (idx < UN + VN + ON) {
            int i = idx - UN - VN; OWbf[i] = f2bf(OW[i]);
        } else {
            int i = idx - UN - VN - ON;
            int k = i >> 8;        // hid row
            int n = i & 255;       // batch col
            HT1[(size_t)n * HID_DIM + k] = f2bf(Hprev[i]);   // transposed
        }
    }
}

// ---------------------------------------------------------------------------
// Double-buffered K=2048 GEMM inner loop over A (row-major bf16) x HT (n-major
// bf16). 4 N-subtiles per wave. Peeled so there are no conditionals in the
// steady state: prefetch (a1,b1) for k+32 while doing WMMAs for k, etc.
// ---------------------------------------------------------------------------
__device__ __forceinline__ void gemm_ht4(const bf16_t* __restrict__ A,
                                         const bf16_t* __restrict__ BT,
                                         int aRow, int nBase, int ll, int lh,
                                         v8f acc[4])
{
    const bf16_t* arow = A + (size_t)aRow * HID_DIM + lh * 16;
    const bf16_t* bcol[4];
#pragma unroll
    for (int j = 0; j < 4; ++j)
        bcol[j] = BT + (size_t)(nBase + j * 16 + ll) * HID_DIM + lh * 16;

    v16bf a0, a1, b0[4], b1[4];
    a0 = loadFragA(arow);
#pragma unroll
    for (int j = 0; j < 4; ++j) b0[j] = loadFragB(bcol[j]);

    int k0 = 0;
    for (; k0 < HID_DIM - 64; k0 += 64) {
        a1 = loadFragA(arow + k0 + 32);
#pragma unroll
        for (int j = 0; j < 4; ++j) b1[j] = loadFragB(bcol[j] + k0 + 32);
#pragma unroll
        for (int j = 0; j < 4; ++j) acc[j] = WMMA_BF16(a0, b0[j], acc[j]);

        a0 = loadFragA(arow + k0 + 64);
#pragma unroll
        for (int j = 0; j < 4; ++j) b0[j] = loadFragB(bcol[j] + k0 + 64);
#pragma unroll
        for (int j = 0; j < 4; ++j) acc[j] = WMMA_BF16(a1, b1[j], acc[j]);
    }
    // tail: k0 == HID_DIM - 64
    a1 = loadFragA(arow + k0 + 32);
#pragma unroll
    for (int j = 0; j < 4; ++j) b1[j] = loadFragB(bcol[j] + k0 + 32);
#pragma unroll
    for (int j = 0; j < 4; ++j) acc[j] = WMMA_BF16(a0, b0[j], acc[j]);
#pragma unroll
    for (int j = 0; j < 4; ++j) acc[j] = WMMA_BF16(a1, b1[j], acc[j]);
}

// ---------------------------------------------------------------------------
// Fused per-step kernel (launch i), grid = 96 blocks of 256 threads:
//   blocks  0..63 : H_i = tanh(U@X_i + V@H_{i-1} + B)   (if doH)  [16 M x 4 N]
//   blocks 64..95 : O_{i-1} = OW@H_{i-1} + OB           (if doO)  [ 8 M x 4 N]
// Both read HTsrc = H_{i-1}; H part writes HTdst = H_i. No intra-launch dep.
// ---------------------------------------------------------------------------
__global__ __launch_bounds__(256)
void rnn_step_kernel(const float*  __restrict__ X,
                     const bf16_t* __restrict__ Ubf,
                     const bf16_t* __restrict__ Vbf,
                     const bf16_t* __restrict__ OWbf,
                     const float*  __restrict__ Bias,
                     const float*  __restrict__ OBias,
                     const bf16_t* __restrict__ HTsrc,
                     bf16_t*       __restrict__ HTdst,
                     float*        __restrict__ Out,
                     int t, int doH, int doO)
{
    __shared__ bf16_t ldsX[64 * 72];            // 64 cols x 64 K, pitch 72 (144B, 16B-aligned)
    const int tid  = threadIdx.x;
    const int lane = tid & 31;
    const int wave = tid >> 5;
    const int lh   = lane >> 4;                 // lane half 0/1
    const int ll   = lane & 15;

    if (blockIdx.x < 64) {
        // ================= H part =================
        if (!doH) return;
        const int mBase = ((int)blockIdx.x >> 2) * 128 + wave * 16;
        const int nBase = ((int)blockIdx.x & 3) * 64;

        v8f acc[4];
#pragma unroll
        for (int j = 0; j < 4; ++j)
#pragma unroll
            for (int r = 0; r < 8; ++r) acc[j][r] = 0.0f;

        const int aRow = mBase + ll;

        // ---- Phase 1: U @ X_t  (X staged fp32->bf16 transposed in LDS, 64-deep) ----
        for (int k0 = 0; k0 < IN_DIM; k0 += 64) {
            {
                int kk = tid >> 2;               // 0..63  (K within chunk)
                int n0 = (tid & 3) * 16;         // 0..48  (col within tile)
                const float4* s4 = (const float4*)(X + (size_t)(k0 + kk) * (SEQ * BATCH)
                                                     + (size_t)t * BATCH + nBase + n0);
#pragma unroll
                for (int q = 0; q < 4; ++q) {
                    float4 v = s4[q];
                    ldsX[(n0 + q * 4 + 0) * 72 + kk] = f2bf(v.x);
                    ldsX[(n0 + q * 4 + 1) * 72 + kk] = f2bf(v.y);
                    ldsX[(n0 + q * 4 + 2) * 72 + kk] = f2bf(v.z);
                    ldsX[(n0 + q * 4 + 3) * 72 + kk] = f2bf(v.w);
                }
            }
            __syncthreads();
#pragma unroll
            for (int kc = 0; kc < 2; ++kc) {
                const bf16_t* ap = Ubf + (size_t)aRow * IN_DIM + k0 + kc * 32 + lh * 16;
                v16bf a = loadFragA(ap);
#pragma unroll
                for (int j = 0; j < 4; ++j) {
                    const bf16_t* bp = ldsX + (j * 16 + ll) * 72 + kc * 32 + lh * 16;
                    v16bf b = loadFragB(bp);
                    acc[j] = WMMA_BF16(a, b, acc[j]);
                }
            }
            __syncthreads();
        }

        // ---- Phase 2: V @ H_{t-1}  (double-buffered, direct from transposed HT) ----
        gemm_ht4(Vbf, HTsrc, aRow, nBase, ll, lh, acc);

        // ---- Epilogue: bias + tanh; store bf16 HT (transposed) [+ fp32 H_last] ----
        const int mRow = mBase + lh * 8;                   // 8 contiguous rows
        v8f bias = *(const v8f*)(Bias + mRow);
        const bool last = (t == SEQ - 1);
#pragma unroll
        for (int j = 0; j < 4; ++j) {
            int n = nBase + j * 16 + ll;
            v8bf hv;
#pragma unroll
            for (int r = 0; r < 8; ++r) {
                float v = tanhf(acc[j][r] + bias[r]);
                acc[j][r] = v;
                hv[r] = f2bf(v);
            }
            *(v8bf*)(HTdst + (size_t)n * HID_DIM + mRow) = hv;
            if (last) {
                float* ho = Out + (size_t)OUT_DIM * SEQ * BATCH + (size_t)mRow * BATCH + n;
#pragma unroll
                for (int r = 0; r < 8; ++r) ho[(size_t)r * BATCH] = acc[j][r];
            }
        }
    } else {
        // ================= O part: O_{t-1} = OW @ H_{t-1} + OB =================
        if (!doO) return;
        const int bb    = (int)blockIdx.x - 64;
        const int mBase = (bb >> 2) * 128 + wave * 16;
        const int nBase = (bb & 3) * 64;

        v8f acc[4];
#pragma unroll
        for (int j = 0; j < 4; ++j)
#pragma unroll
            for (int r = 0; r < 8; ++r) acc[j][r] = 0.0f;

        const int aRow = mBase + ll;
        gemm_ht4(OWbf, HTsrc, aRow, nBase, ll, lh, acc);

        const int mRow = mBase + lh * 8;
        v8f ob = *(const v8f*)(OBias + mRow);
        const int ot = t - 1;
#pragma unroll
        for (int j = 0; j < 4; ++j) {
            int n = nBase + j * 16 + ll;
            float* op = Out + (size_t)mRow * (SEQ * BATCH) + (size_t)ot * BATCH + n;
#pragma unroll
            for (int r = 0; r < 8; ++r) op[(size_t)r * SEQ * BATCH] = acc[j][r] + ob[r];
        }
    }
}

// ---------------------------------------------------------------------------
extern "C" void kernel_launch(void* const* d_in, const int* in_sizes, int n_in,
                              void* d_out, int out_size, void* d_ws, size_t ws_size,
                              hipStream_t stream)
{
    const float* X     = (const float*)d_in[0];
    const float* Hprev = (const float*)d_in[1];
    const float* U     = (const float*)d_in[2];
    const float* V     = (const float*)d_in[3];
    const float* B     = (const float*)d_in[4];
    const float* OW    = (const float*)d_in[5];
    const float* OB    = (const float*)d_in[6];
    float* Out = (float*)d_out;

    char* ws = (char*)d_ws;
    bf16_t* Ubf  = (bf16_t*)(ws);
    bf16_t* Vbf  = (bf16_t*)(ws + (size_t)4  * 1024 * 1024);
    bf16_t* OWbf = (bf16_t*)(ws + (size_t)12 * 1024 * 1024);
    bf16_t* HT0  = (bf16_t*)(ws + (size_t)16 * 1024 * 1024);
    bf16_t* HT1  = (bf16_t*)(ws + (size_t)17 * 1024 * 1024);
    bf16_t* HT[2] = { HT0, HT1 };

    rnn_cvt_kernel<<<2048, 256, 0, stream>>>(U, V, OW, Hprev, Ubf, Vbf, OWbf, HT1);

    // Launch i: computes H_i (i<SEQ) and O_{i-1} (i>=1), pipelined in one kernel.
    // H_i lives in HT[i&1]; H_{-1} (= H_prev) was placed in HT[1] by the cvt kernel.
    for (int i = 0; i <= SEQ; ++i) {
        const bf16_t* src = HT[(i + 1) & 1];   // H_{i-1}
        bf16_t*       dst = HT[i & 1];         // H_i
        rnn_step_kernel<<<96, 256, 0, stream>>>(X, Ubf, Vbf, OWbf, B, OB,
                                                src, dst, Out,
                                                i, (i < SEQ) ? 1 : 0, (i >= 1) ? 1 : 0);
    }
}
